// MultiHeadedAttention_12850542150525
// MI455X (gfx1250) — matrix operand order repaired and checked by reading, not compiled, not run on hardware
//
#include <hip/hip_runtime.h>

#if __has_builtin(__builtin_amdgcn_global_load_async_to_lds_b128) && \
    __has_builtin(__builtin_amdgcn_s_wait_asynccnt)
#define HAS_ASYNC_LDS 1
#else
#define HAS_ASYNC_LDS 0
#endif

typedef __attribute__((ext_vector_type(16))) _Float16 v16h;
typedef __attribute__((ext_vector_type(8)))  _Float16 v8h;
typedef __attribute__((ext_vector_type(8)))  float    v8f;
typedef __attribute__((__vector_size__(4 * sizeof(int)))) int v4i32;

constexpr int Bn  = 4;
constexpr int Sn  = 2048;
constexpr int Dn  = 1024;
constexpr int Hn  = 16;
constexpr int DKn = 64;

// Load halfs [0..7] and [16..23] relative to p (B-fragment K split)
static __device__ __forceinline__ v16h ld_frag_b(const _Float16* p) {
    v8h lo = *(const v8h*)p;
    v8h hi = *(const v8h*)(p + 16);
    return __builtin_shufflevector(lo, hi, 0,1,2,3,4,5,6,7,8,9,10,11,12,13,14,15);
}
// Load halfs [0..7] and [16..23] relative to p (A-fragment K split)
static __device__ __forceinline__ v16h ld_frag_a(const _Float16* p) {
    v8h lo = *(const v8h*)p;
    v8h hi = *(const v8h*)(p + 16);
    return __builtin_shufflevector(lo, hi, 0,1,2,3,4,5,6,7,8,9,10,11,12,13,14,15);
}

#if HAS_ASYNC_LDS
// Per-lane async DMA: LDS[l] = MEM[g] (16 bytes), tracked by ASYNCcnt.
// Builtin signature (from hipcc diagnostic): (v4i32 AS1*, v4i32 AS3*, imm, imm).
static __device__ __forceinline__ void async_b128(const _Float16* g, _Float16* l) {
    __builtin_amdgcn_global_load_async_to_lds_b128(
        (__attribute__((address_space(1))) v4i32*)(unsigned long long)g,
        (__attribute__((address_space(3))) v4i32*)(unsigned)(unsigned long long)l,
        0, 0);
}
static __device__ __forceinline__ void async_wait0() {
    __builtin_amdgcn_s_wait_asynccnt(0);
}
#endif

// ---------------------------------------------------------------------------
// One-time f32 -> f16 conversion (inputs and weights). n8 = n/8.
// ---------------------------------------------------------------------------
__global__ __launch_bounds__(256)
void cvt_f32_to_f16(const float* __restrict__ in, _Float16* __restrict__ out, int n8)
{
    int i = blockIdx.x * 256 + threadIdx.x;
    if (i >= n8) return;
    const float4* p = (const float4*)in + (size_t)i * 2;
    float4 a = p[0], b = p[1];
    v8h o = { (_Float16)a.x, (_Float16)a.y, (_Float16)a.z, (_Float16)a.w,
              (_Float16)b.x, (_Float16)b.y, (_Float16)b.z, (_Float16)b.w };
    *((v8h*)out + i) = o;
}

// ---------------------------------------------------------------------------
// Pack mask [B,S,S] int32 -> bitmask words (bit k of word w = mask[w*32+k]!=0)
// ---------------------------------------------------------------------------
__global__ __launch_bounds__(256)
void pack_mask_kernel(const int* __restrict__ m, unsigned* __restrict__ out, int nwords)
{
    int w = blockIdx.x * 256 + threadIdx.x;
    if (w >= nwords) return;
    const int* p = m + (size_t)w * 32;
    unsigned word = 0;
#pragma unroll
    for (int i = 0; i < 32; ++i) word |= (p[i] != 0 ? 1u : 0u) << i;
    out[w] = word;
}

// ---------------------------------------------------------------------------
// Projection: out = X @ W^T + bias (f16 in/out, f32 accumulate).
// Block = 4 waves sharing a 64x32 W tile (double-buffered, async-to-LDS);
// each wave owns 32(M) x 64(N) -> 8 WMMA per k-step.
// vtrans==0: out[b][h][s][dk] (Q,K).  vtrans==1: out[b][h][dk][s] (V^T).
// ---------------------------------------------------------------------------
__global__ __launch_bounds__(128)
void mha_proj_f16(const _Float16* __restrict__ Xh, const _Float16* __restrict__ Wh,
                  const float* __restrict__ bias, _Float16* __restrict__ out,
                  int vtrans)
{
#if HAS_ASYNC_LDS
    __shared__ _Float16 wtile[2][64 * 32];
#else
    __shared__ _Float16 wtile[1][64 * 32];
#endif
    const int tid  = threadIdx.x;
    const int lane = tid & 31;
    const int wave = tid >> 5;
    const int l16  = lane & 15;
    const int hsel = lane >> 4;

    const int m0      = (blockIdx.x * 4 + wave) * 32;
    const int n_strip = blockIdx.y * 64;

    const _Float16* Xrow0 = Xh + (size_t)(m0 + l16) * Dn;
    const _Float16* Xrow1 = Xh + (size_t)(m0 + 16 + l16) * Dn;
    const int       akoff = hsel * 8;

    const int sa = tid, sb = tid + 128;                 // 16B segments, 256 total
    const size_t gwa = (size_t)(n_strip + (sa >> 2)) * Dn + (sa & 3) * 8;
    const size_t gwb = (size_t)(n_strip + (sb >> 2)) * Dn + (sb & 3) * 8;

    v8f acc0[4] = {}, acc1[4] = {};

#if HAS_ASYNC_LDS
    async_b128(Wh + gwa, &wtile[0][sa * 8]);
    async_b128(Wh + gwb, &wtile[0][sb * 8]);
#endif
    for (int k0 = 0; k0 < Dn; k0 += 32) {
#if HAS_ASYNC_LDS
        const int buf = (k0 >> 5) & 1;
        async_wait0();                     // this tile's DMA done (this wave)
        __syncthreads();                   // visible to block; prev reads done
        if (k0 + 32 < Dn) {                // prefetch next tile into other buf
            async_b128(Wh + gwa + k0 + 32, &wtile[buf ^ 1][sa * 8]);
            async_b128(Wh + gwb + k0 + 32, &wtile[buf ^ 1][sb * 8]);
        }
#else
        const int buf = 0;
        uint4 w0 = *(const uint4*)(Wh + gwa + k0);
        uint4 w1 = *(const uint4*)(Wh + gwb + k0);
        __syncthreads();
        ((uint4*)wtile[0])[sa] = w0;
        ((uint4*)wtile[0])[sb] = w1;
        __syncthreads();
#endif
        v16h a0 = ld_frag_a(Xrow0 + k0 + akoff);
        v16h a1 = ld_frag_a(Xrow1 + k0 + akoff);
#pragma unroll
        for (int t = 0; t < 4; ++t) {
            v16h bf = ld_frag_b(&wtile[buf][(t * 16 + l16) * 32 + hsel * 8]);
            acc0[t] = __builtin_amdgcn_wmma_f32_16x16x32_f16(
                false, a0, false, bf, (short)0, acc0[t], false, false);
            acc1[t] = __builtin_amdgcn_wmma_f32_16x16x32_f16(
                false, a1, false, bf, (short)0, acc1[t], false, false);
        }
    }

    const int bidx  = m0 / Sn;
    const int sbase = m0 % Sn;
#pragma unroll
    for (int t = 0; t < 4; ++t) {
        const int n  = n_strip + t * 16 + l16;
        const int h  = n >> 6;
        const int dk = n & 63;
        const float bv = bias[n];
#pragma unroll
        for (int r = 0; r < 8; ++r) {
            const int s0 = sbase + r + hsel * 8;
            const int s1 = s0 + 16;
            const float val0 = acc0[t][r] + bv;
            const float val1 = acc1[t][r] + bv;
            if (vtrans) {
                size_t base = ((size_t)((bidx * Hn + h) * DKn + dk)) * Sn;
                out[base + s0] = (_Float16)val0;
                out[base + s1] = (_Float16)val1;
            } else {
                size_t base = ((size_t)(bidx * Hn + h)) * Sn * DKn + dk;
                out[base + (size_t)s0 * DKn] = (_Float16)val0;
                out[base + (size_t)s1 * DKn] = (_Float16)val1;
            }
        }
    }
}

// ---------------------------------------------------------------------------
// Flash attention. Qb,Kb: [B,H,S,DK] f16. Vt: [B,H,DK,S] f16. Mp: bit-packed
// mask. Block = 4 waves (64 q rows) sharing double-buffered K/V chunks
// (32 keys) staged via async-to-LDS.
// ---------------------------------------------------------------------------
__global__ __launch_bounds__(128)
void mha_attn_f16(const _Float16* __restrict__ Qb, const _Float16* __restrict__ Kb,
                  const _Float16* __restrict__ Vt, const unsigned* __restrict__ Mp,
                  float* __restrict__ out)
{
#if HAS_ASYNC_LDS
    __shared__ _Float16 ktile[2][32 * 64];     // [key in chunk][d]
    __shared__ _Float16 vtile[2][64 * 32];     // [d][key in chunk]
#else
    __shared__ _Float16 ktile[1][32 * 64];
    __shared__ _Float16 vtile[1][64 * 32];
#endif
    __shared__ _Float16 pstage[4][16 * 32];    // per-wave P transpose

    const int tid  = threadIdx.x;
    const int lane = tid & 31;
    const int wave = tid >> 5;
    const int l16  = lane & 15;
    const int hsel = lane >> 4;

    const int b  = blockIdx.z;
    const int h  = blockIdx.y;
    const int q0 = blockIdx.x * 64 + wave * 16;

    const _Float16* Q = Qb + ((size_t)(b * Hn + h)) * Sn * DKn;
    const _Float16* K = Kb + ((size_t)(b * Hn + h)) * Sn * DKn;
    const _Float16* V = Vt + ((size_t)(b * Hn + h)) * DKn * Sn;
    const unsigned* Mrow = Mp + (size_t)b * Sn * (Sn / 32);

    const int sa = tid, sb = tid + 128;
    const size_t gka = (size_t)(sa >> 3) * DKn + (sa & 7) * 8;  // + j*DKn
    const size_t gkb = (size_t)(sb >> 3) * DKn + (sb & 7) * 8;
    const size_t gva = (size_t)(sa >> 2) * Sn + (sa & 3) * 8;   // + j
    const size_t gvb = (size_t)(sb >> 2) * Sn + (sb & 3) * 8;

    // Resident Q fragments (d = 0..31, 32..63) for rows q0..q0+15
    v16h qf[2];
    {
        const _Float16* Qr = Q + (size_t)(q0 + l16) * DKn;
        qf[0] = ld_frag_a(Qr + hsel * 8);
        qf[1] = ld_frag_a(Qr + 32 + hsel * 8);
    }

    float mrun[8], lrun[8];
#pragma unroll
    for (int r = 0; r < 8; ++r) { mrun[r] = -__builtin_inff(); lrun[r] = 0.f; }
    v8f oacc[4] = {};
    const float scale = 0.125f;                 // 1/sqrt(64)

#if HAS_ASYNC_LDS
    async_b128(K + gka, &ktile[0][sa * 8]);
    async_b128(K + gkb, &ktile[0][sb * 8]);
    async_b128(V + gva, &vtile[0][sa * 8]);
    async_b128(V + gvb, &vtile[0][sb * 8]);
#endif
    for (int j = 0; j < Sn; j += 32) {
#if HAS_ASYNC_LDS
        const int buf = (j >> 5) & 1;
        async_wait0();
        __syncthreads();
        if (j + 32 < Sn) {
            async_b128(K + (size_t)(j + 32) * DKn + gka, &ktile[buf ^ 1][sa * 8]);
            async_b128(K + (size_t)(j + 32) * DKn + gkb, &ktile[buf ^ 1][sb * 8]);
            async_b128(V + gva + j + 32, &vtile[buf ^ 1][sa * 8]);
            async_b128(V + gvb + j + 32, &vtile[buf ^ 1][sb * 8]);
        }
#else
        const int buf = 0;
        uint4 kv0 = *(const uint4*)(K + (size_t)j * DKn + gka);
        uint4 kv1 = *(const uint4*)(K + (size_t)j * DKn + gkb);
        uint4 vv0 = *(const uint4*)(V + gva + j);
        uint4 vv1 = *(const uint4*)(V + gvb + j);
        __syncthreads();
        ((uint4*)ktile[0])[sa] = kv0;  ((uint4*)ktile[0])[sb] = kv1;
        ((uint4*)vtile[0])[sa] = vv0;  ((uint4*)vtile[0])[sb] = vv1;
        __syncthreads();
#endif

        // ---- packed mask words for this chunk -----------------------------
        unsigned mw[8];
#pragma unroll
        for (int r = 0; r < 8; ++r)
            mw[r] = Mrow[(size_t)(q0 + r + hsel * 8) * (Sn / 32) + (j >> 5)];

        // ---- scores: two 16x16 tiles --------------------------------------
        float sc[2][8];
#pragma unroll
        for (int jj = 0; jj < 2; ++jj) {
            v8f st = {};
            const _Float16* Kr = &ktile[buf][(jj * 16 + l16) * 64];
#pragma unroll
            for (int c = 0; c < 2; ++c) {
                v16h bf = ld_frag_b(Kr + c * 32 + hsel * 8);
                st = __builtin_amdgcn_wmma_f32_16x16x32_f16(
                    false, qf[c], false, bf, (short)0, st, false, false);
            }
#pragma unroll
            for (int r = 0; r < 8; ++r) {
                float v = st[r] * scale;
                if (!((mw[r] >> (jj * 16 + l16)) & 1u)) v = -1e9f;
                sc[jj][r] = v;
            }
        }

        // ---- online softmax (row reduce over 16-lane groups) --------------
        float alpha[8];
#pragma unroll
        for (int r = 0; r < 8; ++r) {
            float mx = fmaxf(sc[0][r], sc[1][r]);
            mx = fmaxf(mx, __shfl_xor(mx, 8, 16));
            mx = fmaxf(mx, __shfl_xor(mx, 4, 16));
            mx = fmaxf(mx, __shfl_xor(mx, 2, 16));
            mx = fmaxf(mx, __shfl_xor(mx, 1, 16));
            const float mnew = fmaxf(mrun[r], mx);
            alpha[r] = __expf(mrun[r] - mnew);
            mrun[r]  = mnew;
            const float p0 = __expf(sc[0][r] - mnew);
            const float p1 = __expf(sc[1][r] - mnew);
            sc[0][r] = p0; sc[1][r] = p1;
            float rs = p0 + p1;
            rs += __shfl_xor(rs, 8, 16);
            rs += __shfl_xor(rs, 4, 16);
            rs += __shfl_xor(rs, 2, 16);
            rs += __shfl_xor(rs, 1, 16);
            lrun[r] = lrun[r] * alpha[r] + rs;
        }

        // ---- transpose P (C-layout) -> A-fragment via per-wave LDS --------
        _Float16* stg = pstage[wave];
#pragma unroll
        for (int jj = 0; jj < 2; ++jj)
#pragma unroll
            for (int r = 0; r < 8; ++r)
                stg[(r + hsel * 8) * 32 + jj * 16 + l16] = (_Float16)sc[jj][r];
        __syncthreads();
        v16h pf = ld_frag_a(stg + l16 * 32 + hsel * 8);

        // ---- O = O*alpha + P @ V ------------------------------------------
#pragma unroll
        for (int t = 0; t < 4; ++t) {
#pragma unroll
            for (int r = 0; r < 8; ++r) oacc[t][r] *= alpha[r];
            v16h bf = ld_frag_b(&vtile[buf][(t * 16 + l16) * 32 + hsel * 8]);
            oacc[t] = __builtin_amdgcn_wmma_f32_16x16x32_f16(
                false, pf, false, bf, (short)0, oacc[t], false, false);
        }
    }

    // ---- finalize ---------------------------------------------------------
#pragma unroll
    for (int t = 0; t < 4; ++t)
#pragma unroll
        for (int r = 0; r < 8; ++r) {
            const int row = q0 + r + hsel * 8;
            const int d   = h * DKn + t * 16 + l16;
            out[((size_t)b * Sn + row) * Dn + d] = oacc[t][r] / lrun[r];
        }
}

// ---------------------------------------------------------------------------
extern "C" void kernel_launch(void* const* d_in, const int* in_sizes, int n_in,
                              void* d_out, int out_size, void* d_ws, size_t ws_size,
                              hipStream_t stream)
{
    (void)in_sizes; (void)n_in; (void)out_size; (void)ws_size;
    const float* query = (const float*)d_in[0];
    const float* key   = (const float*)d_in[1];
    const float* value = (const float*)d_in[2];
    const int*   mask  = (const int*)  d_in[3];
    const float* Wq    = (const float*)d_in[4];
    const float* bq    = (const float*)d_in[5];
    const float* Wk    = (const float*)d_in[6];
    const float* bk    = (const float*)d_in[7];
    const float* Wv    = (const float*)d_in[8];
    const float* bv    = (const float*)d_in[9];
    float* out = (float*)d_out;

    const size_t NX  = (size_t)Bn * Sn * Dn;   // 8.39M elements
    const size_t NW  = (size_t)Dn * Dn;        // 1.05M elements
    const size_t NMW = (size_t)Bn * Sn * Sn / 32;

    _Float16* p = (_Float16*)d_ws;
    _Float16 *xq = p, *xk = p + NX, *xv = p + 2 * NX;         p += 3 * NX;
    _Float16 *wqh = p, *wkh = p + NW, *wvh = p + 2 * NW;      p += 3 * NW;
    _Float16 *qbuf = p, *kbuf = p + NX, *vbuf = p + 2 * NX;   p += 3 * NX;
    unsigned* mpack = (unsigned*)p;

    // f32 -> f16 pre-conversion
    cvt_f32_to_f16<<<(int)(NX / 8 / 256), 256, 0, stream>>>(query, xq, (int)(NX / 8));
    cvt_f32_to_f16<<<(int)(NX / 8 / 256), 256, 0, stream>>>(key,   xk, (int)(NX / 8));
    cvt_f32_to_f16<<<(int)(NX / 8 / 256), 256, 0, stream>>>(value, xv, (int)(NX / 8));
    cvt_f32_to_f16<<<(int)(NW / 8 / 256), 256, 0, stream>>>(Wq, wqh, (int)(NW / 8));
    cvt_f32_to_f16<<<(int)(NW / 8 / 256), 256, 0, stream>>>(Wk, wkh, (int)(NW / 8));
    cvt_f32_to_f16<<<(int)(NW / 8 / 256), 256, 0, stream>>>(Wv, wvh, (int)(NW / 8));
    pack_mask_kernel<<<(int)(NMW / 256), 256, 0, stream>>>(mask, mpack, (int)NMW);

    // projections: each wave = 32(M)x64(N), block = 4 waves = 128 M rows
    dim3 pgrid((Bn * Sn / 32) / 4, Dn / 64);                  // (64, 16)
    mha_proj_f16<<<pgrid, 128, 0, stream>>>(xq, wqh, bq, qbuf, 0);
    mha_proj_f16<<<pgrid, 128, 0, stream>>>(xk, wkh, bk, kbuf, 0);
    mha_proj_f16<<<pgrid, 128, 0, stream>>>(xv, wvh, bv, vbuf, 1);

    // attention
    dim3 agrid(Sn / 64, Hn, Bn);                              // (32, 16, 4)
    mha_attn_f16<<<agrid, 128, 0, stream>>>(qbuf, kbuf, vbuf, mpack, out);
}
